// Model_15908558865563
// MI455X (gfx1250) — matrix operand order repaired and checked by reading, not compiled, not run on hardware
//
#include <hip/hip_runtime.h>

typedef __attribute__((ext_vector_type(16))) _Float16 v16h;
typedef __attribute__((ext_vector_type(8)))  float    v8f;
typedef unsigned int uint32x4 __attribute__((ext_vector_type(4)));
typedef int          int32x8  __attribute__((ext_vector_type(8)));
typedef int          int32x4  __attribute__((ext_vector_type(4)));

#define H_DIM 12
#define S_DIM 2048
#define D_DIM 64
#define T_DIM 2048

#define MT 64                 // output rows per workgroup (mask strip height)
#define MASK_LD (T_DIM + 8)   // padded mask row stride in halves
#define NT 128                // phase-2 N tile (columns)
#define KT 64                 // phase-2 K tile
#define XT_LD 72              // x3 stage: [n][k] halves, k-stride pad (64+8)

#if defined(__AMDGCN__) && __has_builtin(__builtin_amdgcn_tensor_load_to_lds) && \
    __has_builtin(__builtin_amdgcn_s_wait_tensorcnt)
#define USE_TDM 1
#else
#define USE_TDM 0
#endif

// --- WMMA f16 fragment index maps (contiguous K, same order for A and B, wave32) ---
__device__ __forceinline__ int ka_idx(int e, int lane) {
  // A 16x32 f16: lane holds row M=lane&15; lanes 0-15: K=0..15, 16-31: K=16..31
  return e + (lane & 16);
}
__device__ __forceinline__ int kb_idx(int e, int lane) {
  // B 32x16 f16: lane holds col N=lane&15; lanes 0-15: K=0..15, 16-31: K=16..31
  return e + (lane & 16);
}

// colsum[u] = sum_t x3[t][u]
__global__ __launch_bounds__(256) void colsum_kernel(const float* __restrict__ x3,
                                                     float* __restrict__ cs) {
  const int u = blockIdx.x * blockDim.x + threadIdx.x;
  float s = 0.0f;
  for (int t = 0; t < T_DIM; ++t) s += x3[(size_t)t * T_DIM + u];
  cs[u] = s;
}

// x3t[u][t] = (f16) x3[t][u]  — LDS-tiled so both global sides are coalesced.
__global__ __launch_bounds__(256) void transpose_f16_kernel(
    const float* __restrict__ x3, _Float16* __restrict__ x3t) {
  __shared__ float tile[64][65];
  const int t0 = (blockIdx.x & 31) * 64;
  const int u0 = (blockIdx.x >> 5) * 64;
  const int tid = threadIdx.x;
#pragma unroll
  for (int i = 0; i < 16; ++i) {
    const int idx = tid + i * 256;
    tile[idx >> 6][idx & 63] = x3[(size_t)(t0 + (idx >> 6)) * T_DIM + (u0 + (idx & 63))];
  }
  __syncthreads();
#pragma unroll
  for (int i = 0; i < 16; ++i) {
    const int idx = tid + i * 256;
    const int ul = idx >> 6, tl = idx & 63;
    x3t[(size_t)(u0 + ul) * T_DIM + (t0 + tl)] = (_Float16)tile[tl][ul];
  }
}

#if USE_TDM
// TDM: DMA a [NT rows x KT halves] f16 tile from x3t (row stride T_DIM halves)
// into LDS at lds_off, inserting 8 halves of pad per row => XT_LD row stride.
__device__ __forceinline__ void tdm_load_tile(unsigned lds_off,
                                              const _Float16* gsrc) {
  const unsigned long long ga = (unsigned long long)(size_t)gsrc;
  uint32x4 g0;
  g0[0] = 1u;                                   // count=1 (valid user D#)
  g0[1] = lds_off;                              // lds_addr
  g0[2] = (unsigned)(ga & 0xffffffffu);         // global_addr[31:0]
  g0[3] = (unsigned)((ga >> 32) & 0x1ffffffu)   // global_addr[56:32]
        | (2u << 30);                           // type=2 ("image")
  int32x8 g1;
  g1[0] = (1 << 16)            // data_size = 2 bytes
        | (1 << 20)            // pad_enable
        | (4 << 22)            // pad_interval: 32 DWORDs (= one 64-half row)
        | (3 << 25);           // pad_amount: 4 DWORDs (= 8 halves)
  g1[1] = (int)((unsigned)T_DIM << 16);  // tensor_dim0 lo16 = 2048
  g1[2] = (int)((unsigned)T_DIM << 16);  // dim0 hi16=0 | tensor_dim1 lo16 = 2048
  g1[3] = (KT << 16);                    // dim1 hi16=0 | tile_dim0 = 64 halves
  g1[4] = NT;                            // tile_dim1 = 128 rows, tile_dim2 = 0
  g1[5] = T_DIM;                         // tensor_dim0_stride = 2048 (lo32)
  g1[6] = 0;
  g1[7] = 0;
  const int32x4 z4 = {0, 0, 0, 0};             // groups 2/3 unused (2D tensor)
  const int32x8 z8 = {0, 0, 0, 0, 0, 0, 0, 0}; // extra group operand (unused)
  __builtin_amdgcn_tensor_load_to_lds(g0, g1, z4, z4, z8, 0);
}
#endif

// One K-step of GEMM2: acc[0..3] += mask[16 rows x 64K] @ xtb[64K x 64N]
__device__ __forceinline__ void gemm2_step(v8f acc[4], const _Float16* __restrict__ maskS,
                                           const _Float16* __restrict__ xtb, int k0,
                                           int lrow_base, int ln16, int lane, int wn) {
  v16h ma0, ma1;
  const _Float16* mrow = maskS + (size_t)(lrow_base + ln16) * MASK_LD + k0;
#pragma unroll
  for (int e = 0; e < 16; ++e) {
    ma0[e] = mrow[ka_idx(e, lane)];
    ma1[e] = mrow[32 + ka_idx(e, lane)];
  }
#pragma unroll
  for (int f = 0; f < 4; ++f) {
    const _Float16* xrow = xtb + (size_t)(wn * 64 + f * 16 + ln16) * XT_LD;
    v16h b0, b1;
#pragma unroll
    for (int e = 0; e < 16; ++e) {
      b0[e] = xrow[kb_idx(e, lane)];
      b1[e] = xrow[32 + kb_idx(e, lane)];
    }
    acc[f] = __builtin_amdgcn_wmma_f32_16x16x32_f16(false, ma0, false, b0, (short)0, acc[f], false, false);
    acc[f] = __builtin_amdgcn_wmma_f32_16x16x32_f16(false, ma1, false, b1, (short)0, acc[f], false, false);
  }
}

template <bool USE_XT>
__global__ __launch_bounds__(256) void fused_mask_attn(
    const float* __restrict__ x1, const float* __restrict__ x2,
    const float* __restrict__ x3, const _Float16* __restrict__ x3t,
    const float* __restrict__ colsum, float* __restrict__ out) {
  extern __shared__ char smem_raw[];
  _Float16* maskS = (_Float16*)smem_raw;                              // [MT][MASK_LD]
  _Float16* xt    = (_Float16*)(smem_raw + (size_t)MT * MASK_LD * 2); // 2 x [NT][XT_LD]
  float* kcount   = (float*)(smem_raw + (size_t)MT * MASK_LD * 2 + (size_t)2 * NT * XT_LD * 2);
  float* invden   = kcount + MT;

  const int tid  = threadIdx.x;
  const int lane = tid & 31;
  const int wave = tid >> 5;   // 8 waves: 4 (M) x 2 (N)
  const int wm   = wave >> 1;
  const int wn   = wave & 1;
  const int ln16 = lane & 15;
  const int hi   = (lane >> 4) & 1;

  const int h  = blockIdx.x >> 5;        // S_DIM/MT == 32 blocks per head
  const int mb = blockIdx.x & 31;
  const int m0 = mb * MT;
  const int lrow_base = wm * 16;

  if (tid < MT) kcount[tid] = 0.0f;
  __syncthreads();

  const float SCALE = 0.022097086912079608f;  // 1/sqrt(2048)
  const float EUL   = 2.7182818284590452f;
  const float EM1   = 1.7182818284590452f;

  // ================= Phase 1: mask = (x1@x2^T * scale + x3 <= 0) ============
  const float* arow = x1 + ((size_t)h * S_DIM + (size_t)(m0 + lrow_base + ln16)) * D_DIM;
  v16h a0, a1;
#pragma unroll
  for (int e = 0; e < 16; ++e) {
    a0[e] = (_Float16)arow[ka_idx(e, lane)];
    a1[e] = (_Float16)arow[32 + ka_idx(e, lane)];
  }

  float kacc[8];
#pragma unroll
  for (int r = 0; r < 8; ++r) kacc[r] = 0.0f;

  for (int j = 0; j < T_DIM / 32; ++j) {
    const int t0 = wn * 16 + j * 32;
    const float* brow = x2 + (size_t)(t0 + ln16) * D_DIM;
    v16h b0, b1;
#pragma unroll
    for (int e = 0; e < 16; ++e) {
      b0[e] = (_Float16)brow[kb_idx(e, lane)];
      b1[e] = (_Float16)brow[32 + kb_idx(e, lane)];
    }
    v8f c = {};
    c = __builtin_amdgcn_wmma_f32_16x16x32_f16(false, a0, false, b0, (short)0, c, false, false);
    c = __builtin_amdgcn_wmma_f32_16x16x32_f16(false, a1, false, b1, (short)0, c, false, false);

    const int tcol = t0 + ln16;
#pragma unroll
    for (int r = 0; r < 8; ++r) {
      const int lrow = lrow_base + r + hi * 8;
      const float v = c[r] * SCALE + x3[(size_t)(m0 + lrow) * T_DIM + tcol];
      const float m = (v <= 0.0f) ? 1.0f : 0.0f;
      kacc[r] += m;
      maskS[(size_t)lrow * MASK_LD + tcol] = (_Float16)m;
    }
  }
#pragma unroll
  for (int r = 0; r < 8; ++r)
    atomicAdd(&kcount[lrow_base + r + hi * 8], kacc[r]);   // ds_add_f32
  __syncthreads();

  if (tid < MT) {
    const float k = kcount[tid];
    invden[tid] = 1.0f / (k * EUL + ((float)T_DIM - k));
  }
  __syncthreads();

  float invr[8];
#pragma unroll
  for (int r = 0; r < 8; ++r) invr[r] = invden[lrow_base + r + hi * 8];

  // ====== Phase 2: out = invden * (colsum(x3) + (e-1) * (mask @ x3)) ========
  _Float16* xtb0 = xt;
  _Float16* xtb1 = xt + (size_t)NT * XT_LD;

  for (int nt = 0; nt < T_DIM / NT; ++nt) {
    const int n0 = nt * NT;
    v8f acc[4] = {{}, {}, {}, {}};

    if constexpr (USE_XT) {
#if USE_TDM
      // --- TDM double-buffered DMA pipeline (TENSORcnt) ---
      if (wave == 0)
        tdm_load_tile((unsigned)(size_t)xtb0, x3t + (size_t)n0 * T_DIM);
      int cur = 0;
      for (int k0 = 0; k0 < T_DIM; k0 += KT) {
        if (wave == 0) {
          if (k0 + KT < T_DIM) {
            tdm_load_tile((unsigned)(size_t)(cur ? xtb0 : xtb1),
                          x3t + (size_t)n0 * T_DIM + (k0 + KT));
            __builtin_amdgcn_s_wait_tensorcnt((short)1);  // current tile done
          } else {
            __builtin_amdgcn_s_wait_tensorcnt((short)0);
          }
        }
        __syncthreads();
        gemm2_step(acc, maskS, cur ? xtb1 : xtb0, k0, lrow_base, ln16, lane, wn);
        __syncthreads();
        cur ^= 1;
      }
#else
      // --- vectorized b128 copy from pre-converted f16 x3t ---
      for (int k0 = 0; k0 < T_DIM; k0 += KT) {
#pragma unroll
        for (int i = 0; i < (NT * KT) / (8 * 256); ++i) {  // 1024 uint4 chunks
          const int c  = tid + i * 256;
          const int nn = c >> 3;
          const int k8 = (c & 7) * 8;
          *(uint4*)(xtb0 + (size_t)nn * XT_LD + k8) =
              *(const uint4*)(x3t + (size_t)(n0 + nn) * T_DIM + k0 + k8);
        }
        __syncthreads();
        gemm2_step(acc, maskS, xtb0, k0, lrow_base, ln16, lane, wn);
        __syncthreads();
      }
#endif
    } else {
      // --- fallback: stage from f32 x3 with on-the-fly conversion ---
      for (int k0 = 0; k0 < T_DIM; k0 += KT) {
#pragma unroll
        for (int i = 0; i < (KT * NT) / 256; ++i) {
          const int idx = tid + i * 256;
          const int kk = idx >> 7;
          const int nn = idx & 127;
          const size_t gofs = (size_t)(k0 + kk) * T_DIM + (n0 + nn);
          xtb0[(size_t)nn * XT_LD + kk] = (_Float16)x3[gofs];
          if (k0 + KT < T_DIM)
            __builtin_prefetch(&x3[gofs + (size_t)KT * T_DIM], 0, 1);
        }
        __syncthreads();
        gemm2_step(acc, maskS, xtb0, k0, lrow_base, ln16, lane, wn);
        __syncthreads();
      }
    }

    // Epilogue
#pragma unroll
    for (int f = 0; f < 4; ++f) {
      const int u = n0 + wn * 64 + f * 16 + ln16;
      const float cs = colsum[u];
#pragma unroll
      for (int r = 0; r < 8; ++r) {
        const int srow = m0 + lrow_base + r + hi * 8;
        out[((size_t)h * S_DIM + srow) * T_DIM + u] = invr[r] * (cs + EM1 * acc[f][r]);
      }
    }
  }
}

extern "C" void kernel_launch(void* const* d_in, const int* in_sizes, int n_in,
                              void* d_out, int out_size, void* d_ws, size_t ws_size,
                              hipStream_t stream) {
  const float* x1 = (const float*)d_in[0];  // [1,12,2048,64] f32
  const float* x2 = (const float*)d_in[1];  // [2048,64]      f32
  const float* x3 = (const float*)d_in[2];  // [2048,2048]    f32
  float* out = (float*)d_out;
  float* cs  = (float*)d_ws;                // colsum: 8 KB

  colsum_kernel<<<T_DIM / 256, 256, 0, stream>>>(x3, cs);

  const size_t shmem = (size_t)MT * MASK_LD * 2        // mask strip (~257 KB)
                     + (size_t)2 * NT * XT_LD * 2      // double-buffered x3 tile (36 KB)
                     + (size_t)2 * MT * 4;             // kcount + invden
  const size_t csBytes  = 8192;
  const size_t x3tBytes = (size_t)T_DIM * T_DIM * sizeof(_Float16);  // 8 MB, L2-resident

  if (ws_size >= csBytes + x3tBytes) {
    _Float16* x3t = (_Float16*)((char*)d_ws + csBytes);
    transpose_f16_kernel<<<32 * 32, 256, 0, stream>>>(x3, x3t);
    fused_mask_attn<true><<<H_DIM * (S_DIM / MT), 256, shmem, stream>>>(
        x1, x2, x3, x3t, cs, out);
  } else {
    fused_mask_attn<false><<<H_DIM * (S_DIM / MT), 256, shmem, stream>>>(
        x1, x2, x3, nullptr, cs, out);
  }
}
